// ResEncoderBlock_23098334118048
// MI455X (gfx1250) — matrix operand order repaired and checked by reading, not compiled, not run on hardware
//
#include <hip/hip_runtime.h>
#include <hip/hip_bf16.h>
#include <math.h>

// Problem constants (reference: B,T,H,S = 2,2048,16,64; E=1024; FF=4096)
#define BQ 2
#define TT 2048
#define HH 16
#define SS 64
#define EE 1024
#define FFN 4096

typedef __attribute__((ext_vector_type(16))) __bf16 v16bf;
typedef __attribute__((ext_vector_type(8)))  float  v8f;

union FragAB { v16bf v; uint4 q[2]; unsigned short u[16]; };
union FragC  { v8f v; float e[8]; };

// float -> bf16 bits, round-to-nearest-even
__device__ __forceinline__ unsigned short f2bf(float f) {
  unsigned u = __float_as_uint(f);
  u += 0x7FFFu + ((u >> 16) & 1u);
  return (unsigned short)(u >> 16);
}

__device__ __forceinline__ v8f wmma_bf16(v16bf a, v16bf b, v8f c) {
  // D = A(16x32 bf16) * B(32x16 bf16) + C(16x16 f32)
  return __builtin_amdgcn_wmma_f32_16x16x32_bf16(false, a, false, b, (short)0, c,
                                                 false, false);
}

// ---- CDNA5 async copy: global -> LDS, 16B per lane, tracked by ASYNCcnt ----
__device__ __forceinline__ void async_copy_b128(unsigned lds_addr, const void* gaddr) {
  asm volatile("global_load_async_to_lds_b128 %0, %1, off"
               :: "v"(lds_addr), "v"(gaddr) : "memory");
}
__device__ __forceinline__ void wait_asynccnt0() {
  asm volatile("s_wait_asynccnt 0x0" ::: "memory");
}
// Low 32 bits of a generic pointer into LDS are the LDS byte offset
// (ISA 10.2, LDS aperture).
__device__ __forceinline__ unsigned lds_off(const void* p) {
  return (unsigned)(unsigned long long)p;
}

// A fragment (16x32 bf16). `row` points at this lane's row at the k-chunk start.
// Lane holds m = lane&15; k in {base..base+7} U {16+base..16+base+7},
// base = 8*(lane>>4). Two contiguous 16B loads.
__device__ __forceinline__ v16bf load_a_bf(const unsigned short* row, int lane) {
  const int base = (lane >> 4) << 3;
  FragAB f;
  f.q[0] = *(const uint4*)(row + base);
  f.q[1] = *(const uint4*)(row + 16 + base);
  return f.v;
}

// A fragment built from an fp32 row (converted on the fly).
__device__ __forceinline__ v16bf load_a_f32(const float* row, int lane) {
  const int base = (lane >> 4) << 3;
  FragAB f;
#pragma unroll
  for (int g = 0; g < 2; ++g) {
    const float* p = row + base + g * 16;
    float4 a = *(const float4*)(p);
    float4 b = *(const float4*)(p + 4);
    int o = g * 8;
    f.u[o + 0] = f2bf(a.x); f.u[o + 1] = f2bf(a.y);
    f.u[o + 2] = f2bf(a.z); f.u[o + 3] = f2bf(a.w);
    f.u[o + 4] = f2bf(b.x); f.u[o + 5] = f2bf(b.y);
    f.u[o + 6] = f2bf(b.z); f.u[o + 7] = f2bf(b.w);
  }
  return f.v;
}

// B fragment (32x16 bf16). Lane holds n = lane&15; k in {base..base+7} U
// {16+base..16+base+7}, base = 8*(lane>>4); p points at B[k0][n0], row stride ld.
__device__ __forceinline__ v16bf load_b_bf(const unsigned short* p, int ld, int lane) {
  const int base = (lane >> 4) << 3;
  FragAB f;
#pragma unroll
  for (int s = 0; s < 8; ++s) {
    f.u[s]     = p[(size_t)(base + s) * ld + (lane & 15)];
    f.u[8 + s] = p[(size_t)(16 + base + s) * ld + (lane & 15)];
  }
  return f.v;
}

// ---------------------------------------------------------------- weights
// Wt[k*N + n] = bf16(W[n*K + k])   (B-operand layout: row k contiguous in n)
__global__ void k_transpose_cvt(const float* __restrict__ W,
                                unsigned short* __restrict__ Wt, int N, int K) {
  int idx = blockIdx.x * blockDim.x + threadIdx.x;
  if (idx >= N * K) return;
  int n = idx % N, k = idx / N;
  Wt[(size_t)k * N + n] = f2bf(W[(size_t)n * K + k]);
}

// ---------------------------------------------------------------- KQV
// out[m, o] = sum_s x[m*64 + s] * kqv_w[o, s],  m = (b*T + t)*H + h (65536 rows)
// region nt>>2: 0 -> K (store transposed [b,h,s,t]); 1 -> Q (scaled 1/8,
// [b,h,t,s]); 2 -> V ([b,t,h,s]); all bf16. Region branch is wave-uniform.
__global__ void __launch_bounds__(256) k_kqv(const float* __restrict__ x,
                                             const unsigned short* __restrict__ WkT,
                                             unsigned short* __restrict__ Kt,
                                             unsigned short* __restrict__ Qs,
                                             unsigned short* __restrict__ Vb) {
  const int lane = threadIdx.x & 31;
  const int wave = blockIdx.x * (blockDim.x >> 5) + (threadIdx.x >> 5);
  const int NT = 192 / 16;
  const int nt = wave % NT;
  const int mt = wave / NT;
  const int m0 = mt * 16;
  const float* arow = x + (size_t)(m0 + (lane & 15)) * SS;
  v8f acc = {};
#pragma unroll
  for (int c = 0; c < 2; ++c) {
    v16bf a = load_a_f32(arow + c * 32, lane);
    v16bf b = load_b_bf(WkT + (size_t)(c * 32) * 192 + nt * 16, 192, lane);
    acc = wmma_bf16(a, b, acc);
  }
  FragC d; d.v = acc;
  const int col = nt * 16 + (lane & 15);
  const int region = nt >> 2;  // wave-uniform
  int mm[8], hh[8], tt[8], bb[8];
#pragma unroll
  for (int r = 0; r < 8; ++r) {
    mm[r] = m0 + r + ((lane >> 4) << 3);
    hh[r] = mm[r] & (HH - 1);
    tt[r] = (mm[r] >> 4) & (TT - 1);
    bb[r] = mm[r] >> 15;  // (m>>4)>>11
  }
  if (region == 0) {
#pragma unroll
    for (int r = 0; r < 8; ++r)
      Kt[((size_t)(bb[r] * HH + hh[r]) * SS + col) * TT + tt[r]] = f2bf(d.e[r]);
  } else if (region == 1) {
#pragma unroll
    for (int r = 0; r < 8; ++r)
      Qs[((size_t)(bb[r] * HH + hh[r]) * TT + tt[r]) * SS + (col - 64)] =
          f2bf(d.e[r] * 0.125f);
  } else {
#pragma unroll
    for (int r = 0; r < 8; ++r)
      Vb[((size_t)(bb[r] * TT + tt[r]) * HH + hh[r]) * SS + (col - 128)] =
          f2bf(d.e[r]);
  }
}

// ---------------------------------------------------------------- scores
// att[b,i,j,h] = (Q/8)[b,h,i,:] . K[b,h,:,j] + prev[b,i,j,h]
// prev is fed straight in as the WMMA C accumulator. prev/att are streamed
// (537MB each, >> L2) -> non-temporal.
__global__ void __launch_bounds__(256) k_scores(const unsigned short* __restrict__ Qs,
                                                const unsigned short* __restrict__ Kt,
                                                const float* __restrict__ prev,
                                                float* __restrict__ att) {
  const int lane = threadIdx.x & 31;
  const int wave = blockIdx.x * (blockDim.x >> 5) + (threadIdx.x >> 5);
  const int jt = wave & 127;
  int w2 = wave >> 7;
  const int it = w2 & 127;
  w2 >>= 7;
  const int h = w2 & (HH - 1);
  const int b = w2 >> 4;
  const int i0 = it * 16, j0 = jt * 16;
  const unsigned short* qrow =
      Qs + ((size_t)(b * HH + h) * TT + (i0 + (lane & 15))) * SS;
  const int j = j0 + (lane & 15);
  FragC c;
#pragma unroll
  for (int r = 0; r < 8; ++r) {
    int i = i0 + r + ((lane >> 4) << 3);
    c.e[r] = __builtin_nontemporal_load(
        prev + ((size_t)(b * TT + i) * TT + j) * HH + h);
  }
  v8f acc = c.v;
#pragma unroll
  for (int cc = 0; cc < 2; ++cc) {
    v16bf a = load_a_bf(qrow + cc * 32, lane);
    v16bf bb = load_b_bf(Kt + ((size_t)(b * HH + h) * SS + cc * 32) * TT + j0, TT, lane);
    acc = wmma_bf16(a, bb, acc);
  }
  FragC d; d.v = acc;
#pragma unroll
  for (int r = 0; r < 8; ++r) {
    int i = i0 + r + ((lane >> 4) << 3);
    __builtin_nontemporal_store(d.e[r],
        att + ((size_t)(b * TT + i) * TT + j) * HH + h);
  }
}

// ---------------------------------------------------------------- softmax stats
// One wave per (b,i,h) row: online max + sum(exp), then lane-shuffle combine.
__global__ void __launch_bounds__(256) k_rowstats(const float* __restrict__ att,
                                                  float* __restrict__ stat_m,
                                                  float* __restrict__ stat_inv) {
  const int lane = threadIdx.x & 31;
  const int row = blockIdx.x * (blockDim.x >> 5) + (threadIdx.x >> 5); // (b*T+i)*H+h
  const int h = row & (HH - 1);
  const int bt = row >> 4;
  const float* base = att + (size_t)bt * TT * HH + h;
  float m = -3.0e38f, s = 0.f;
  for (int j = lane; j < TT; j += 32) {
    __builtin_prefetch(base + (size_t)(j + 128) * HH, 0, 0);
    float v = base[(size_t)j * HH];
    float nm = fmaxf(m, v);
    s = s * __expf(m - nm) + __expf(v - nm);
    m = nm;
  }
#pragma unroll
  for (int off = 16; off > 0; off >>= 1) {
    float om = __shfl_xor(m, off);
    float os = __shfl_xor(s, off);
    float nm = fmaxf(m, om);
    s = s * __expf(m - nm) + os * __expf(om - nm);
    m = nm;
  }
  if (lane == 0) { stat_m[row] = m; stat_inv[row] = 1.0f / s; }
}

// ---------------------------------------------------------------- P @ V
// Block = 8 waves, all same (b,h), consecutive i-tiles. The 32-row V chunk
// (4KB) is shared by all 8 waves: wave 0 async-stages it into LDS
// (double-buffered), everyone reads B fragments from LDS.
__global__ void __launch_bounds__(256) k_pv(const float* __restrict__ att,
                                            const unsigned short* __restrict__ Vb,
                                            const float* __restrict__ stat_m,
                                            const float* __restrict__ stat_inv,
                                            unsigned short* __restrict__ res_bf) {
  __shared__ __align__(16) unsigned short sV[2][32 * SS];  // 2 x 4KB
  const int lane = threadIdx.x & 31;
  const int wid  = threadIdx.x >> 5;
  const int itb = blockIdx.x & 15;        // 16 blocks cover 128 i-tiles
  const int bh  = blockIdx.x >> 4;
  const int h = bh & (HH - 1);
  const int b = bh >> 4;
  const int it = itb * 8 + wid;
  const int i0 = it * 16;
  const int irow = i0 + (lane & 15);
  const float mi = stat_m[(size_t)(b * TT + irow) * HH + h];
  const float* srow = att + (size_t)(b * TT + irow) * TT * HH + h;
  const int base = (lane >> 4) << 3;
  // V row for async staging: lane L stages key-row j = c*32 + L (128B = 8x16B)
  const unsigned short* vlane =
      Vb + ((size_t)(b * TT + lane) * HH + h) * SS;  // + c*32*HH*SS per chunk
  const size_t vstep = (size_t)32 * HH * SS;
  if (wid == 0) {
    unsigned l0 = lds_off(&sV[0][lane * SS]);
#pragma unroll
    for (int q = 0; q < 8; ++q) async_copy_b128(l0 + q * 16, vlane + q * 8);
    wait_asynccnt0();
  }
  __syncthreads();
  v8f acc0 = {}, acc1 = {}, acc2 = {}, acc3 = {};
  const int C = TT / 32;
  for (int c = 0; c < C; ++c) {
    const int cur = c & 1;
    if (wid == 0 && c + 1 < C) {
      unsigned l0 = lds_off(&sV[cur ^ 1][lane * SS]);
      const unsigned short* g = vlane + (size_t)(c + 1) * vstep;
#pragma unroll
      for (int q = 0; q < 8; ++q) async_copy_b128(l0 + q * 16, g + q * 8);
    }
    const int j0 = c * 32;
    FragAB a;
#pragma unroll
    for (int i = 0; i < 8; ++i) {
      __builtin_prefetch(srow + (size_t)(j0 + 32 + base + i) * HH, 0, 0);
      float v0 = srow[(size_t)(j0 + base + i) * HH];
      float v1 = srow[(size_t)(j0 + 16 + base + i) * HH];
      a.u[i]     = f2bf(__expf(v0 - mi));
      a.u[8 + i] = f2bf(__expf(v1 - mi));
    }
    const unsigned short* vrow = &sV[cur][0];  // LDS: B rows k, stride SS
    acc0 = wmma_bf16(a.v, load_b_bf(vrow, SS, lane),      acc0);
    acc1 = wmma_bf16(a.v, load_b_bf(vrow + 16, SS, lane), acc1);
    acc2 = wmma_bf16(a.v, load_b_bf(vrow + 32, SS, lane), acc2);
    acc3 = wmma_bf16(a.v, load_b_bf(vrow + 48, SS, lane), acc3);
    if (wid == 0) wait_asynccnt0();
    __syncthreads();
  }
  float inv[8];
#pragma unroll
  for (int r = 0; r < 8; ++r) {
    int i = i0 + r + ((lane >> 4) << 3);
    inv[r] = stat_inv[(size_t)(b * TT + i) * HH + h];
  }
  v8f accs[4] = {acc0, acc1, acc2, acc3};
  const int ncol = lane & 15;
#pragma unroll
  for (int nt = 0; nt < 4; ++nt) {
    FragC d; d.v = accs[nt];
#pragma unroll
    for (int r = 0; r < 8; ++r) {
      int i = i0 + r + ((lane >> 4) << 3);
      res_bf[(size_t)(b * TT + i) * EE + h * SS + nt * 16 + ncol] =
          f2bf(d.e[r] * inv[r]);
    }
  }
}

// ---------------------------------------------------------------- generic GEMM
// out[m,n] = act( A[m,:].Bt[:,n] + bias[n] );  A bf16 MxK, Bt bf16 KxN.
// Block = 8 waves = 8 m-tiles sharing one n-tile; the shared 1KB B chunk is
// async-staged into LDS by wave 0 (double-buffered).
__global__ void __launch_bounds__(256) k_gemm(const unsigned short* __restrict__ A,
                                              const unsigned short* __restrict__ Bt,
                                              const float* __restrict__ bias,
                                              float* __restrict__ outf,
                                              unsigned short* __restrict__ outbf,
                                              int M, int N, int K, int gelu) {
  __shared__ __align__(16) unsigned short sB[2][32 * 16];  // 2 x 1KB
  const int lane = threadIdx.x & 31;
  const int wid  = threadIdx.x >> 5;
  const int NT = N >> 4;
  const int nt = blockIdx.x % NT;
  const int mb = blockIdx.x / NT;
  const int mt = mb * 8 + wid;
  const unsigned short* arow = A + (size_t)(mt * 16 + (lane & 15)) * K;
  const unsigned short* bcol = Bt + (size_t)lane * N + nt * 16;  // row k = lane
  if (wid == 0) {
    unsigned l0 = lds_off(&sB[0][lane * 16]);
    async_copy_b128(l0, bcol);
    async_copy_b128(l0 + 16, bcol + 8);
    wait_asynccnt0();
  }
  __syncthreads();
  v8f acc = {};
  const int C = K >> 5;
  for (int c = 0; c < C; ++c) {
    const int cur = c & 1;
    if (wid == 0 && c + 1 < C) {
      unsigned l0 = lds_off(&sB[cur ^ 1][lane * 16]);
      const unsigned short* g = bcol + (size_t)(c + 1) * 32 * N;
      async_copy_b128(l0, g);
      async_copy_b128(l0 + 16, g + 8);
    }
    v16bf a = load_a_bf(arow + c * 32, lane);
    v16bf b = load_b_bf(&sB[cur][0], 16, lane);
    acc = wmma_bf16(a, b, acc);
    if (wid == 0) wait_asynccnt0();
    __syncthreads();
  }
  FragC d; d.v = acc;
  const int n = nt * 16 + (lane & 15);
  const float bv = bias ? bias[n] : 0.0f;
#pragma unroll
  for (int r = 0; r < 8; ++r) {
    int m = mt * 16 + r + ((lane >> 4) << 3);
    float v = d.e[r] + bv;
    if (gelu) v = 0.5f * v * (1.0f + erff(v * 0.70710678f));
    if (outf)  outf[(size_t)m * N + n] = v;
    if (outbf) outbf[(size_t)m * N + n] = f2bf(v);
  }
}

// ---------------------------------------------------------------- add + LayerNorm
__device__ __forceinline__ float block_sum(float val, float* red) {
  int lane = threadIdx.x & 31, wid = threadIdx.x >> 5;
#pragma unroll
  for (int o = 16; o > 0; o >>= 1) val += __shfl_xor(val, o);
  if (lane == 0) red[wid] = val;
  __syncthreads();
  if (wid == 0) {
    float t = (lane < (int)(blockDim.x >> 5)) ? red[lane] : 0.f;
#pragma unroll
    for (int o = 16; o > 0; o >>= 1) t += __shfl_xor(t, o);
    if (lane == 0) red[0] = t;
  }
  __syncthreads();
  float r = red[0];
  __syncthreads();
  return r;
}

__global__ void __launch_bounds__(256) k_add_ln(const float* __restrict__ xa,
                                                const float* __restrict__ xb,
                                                const float* __restrict__ w,
                                                const float* __restrict__ bias,
                                                float* __restrict__ outf,
                                                unsigned short* __restrict__ outbf) {
  __shared__ float red[32];
  const int row = blockIdx.x;
  const int tid = threadIdx.x;
  const float* pa = xa + (size_t)row * EE;
  const float* pb = xb + (size_t)row * EE;
  float v[4];
  float s = 0.f;
#pragma unroll
  for (int i = 0; i < 4; ++i) {
    int e = tid + i * 256;
    v[i] = pa[e] + pb[e];
    s += v[i];
  }
  float mu = block_sum(s, red) * (1.0f / EE);
  float q = 0.f;
#pragma unroll
  for (int i = 0; i < 4; ++i) { float dd = v[i] - mu; q += dd * dd; }
  float var = block_sum(q, red) * (1.0f / EE);
  float rs = rsqrtf(var + 1e-5f);
#pragma unroll
  for (int i = 0; i < 4; ++i) {
    int e = tid + i * 256;
    float y = (v[i] - mu) * rs * w[e] + bias[e];
    if (outf)  outf[(size_t)row * EE + e] = y;
    if (outbf) outbf[(size_t)row * EE + e] = f2bf(y);
  }
}

// ---------------------------------------------------------------- launch
extern "C" void kernel_launch(void* const* d_in, const int* in_sizes, int n_in,
                              void* d_out, int out_size, void* d_ws, size_t ws_size,
                              hipStream_t stream) {
  const float* x      = (const float*)d_in[0];
  const float* prev   = (const float*)d_in[1];
  const float* kqv_w  = (const float*)d_in[2];
  const float* proj_w = (const float*)d_in[3];
  const float* ln1_w  = (const float*)d_in[4];
  const float* ln1_b  = (const float*)d_in[5];
  const float* ln2_w  = (const float*)d_in[6];
  const float* ln2_b  = (const float*)d_in[7];
  const float* ff_w1  = (const float*)d_in[8];
  const float* ff_b1  = (const float*)d_in[9];
  const float* ff_w2  = (const float*)d_in[10];
  const float* ff_b2  = (const float*)d_in[11];

  float* x2_out  = (float*)d_out;                         // [B,T,E]
  float* att_out = (float*)d_out + (size_t)BQ * TT * EE;  // [B,T,T,H]

  char* p = (char*)d_ws;
  auto take = [&](size_t bytes) -> void* {
    void* r = (void*)p;
    p += (bytes + 255) & ~(size_t)255;
    return r;
  };
  const size_t nBTHS = (size_t)BQ * TT * HH * SS;  // 4,194,304
  unsigned short* Kt     = (unsigned short*)take(nBTHS * 2);       // [b,h,s,t]
  unsigned short* Qs     = (unsigned short*)take(nBTHS * 2);       // [b,h,t,s] (pre-scaled)
  unsigned short* Vb     = (unsigned short*)take(nBTHS * 2);       // [b,t,h,s]
  float*          stat_m = (float*)take((size_t)BQ * TT * HH * 4);
  float*          stat_i = (float*)take((size_t)BQ * TT * HH * 4);
  unsigned short* WkT    = (unsigned short*)take((size_t)SS * 192 * 2);
  unsigned short* projT  = (unsigned short*)take((size_t)EE * EE * 2);
  unsigned short* ff1T   = (unsigned short*)take((size_t)EE * FFN * 2);
  unsigned short* ff2T   = (unsigned short*)take((size_t)FFN * EE * 2);
  unsigned short* res_bf = (unsigned short*)take((size_t)BQ * TT * EE * 2);
  float*          mha    = (float*)take((size_t)BQ * TT * EE * 4);
  float*          x1f    = (float*)take((size_t)BQ * TT * EE * 4);
  unsigned short* x1bf   = (unsigned short*)take((size_t)BQ * TT * EE * 2);
  unsigned short* h1bf   = (unsigned short*)take((size_t)BQ * TT * FFN * 2);
  float*          fff    = (float*)take((size_t)BQ * TT * EE * 4);

  // 1) weight transposes to bf16 B-operand layout
  k_transpose_cvt<<<(192 * 64 + 255) / 256, 256, 0, stream>>>(kqv_w, WkT, 192, 64);
  k_transpose_cvt<<<(EE * EE + 255) / 256, 256, 0, stream>>>(proj_w, projT, EE, EE);
  k_transpose_cvt<<<(FFN * EE + 255) / 256, 256, 0, stream>>>(ff_w1, ff1T, FFN, EE);
  k_transpose_cvt<<<(FFN * EE + 255) / 256, 256, 0, stream>>>(ff_w2, ff2T, EE, FFN);

  // 2) KQV projection (M=65536, N=192, K=64)
  k_kqv<<<(4096 * 12) / 8, 256, 0, stream>>>(x, WkT, Kt, Qs, Vb);

  // 3) attention scores (+prev) -> att_out   (2*16*128*128 wave-tiles)
  k_scores<<<(BQ * HH * 128 * 128) / 8, 256, 0, stream>>>(Qs, Kt, prev, att_out);

  // 4) softmax row stats (one wave per (b,i,h) row)
  k_rowstats<<<(BQ * TT * HH) / 8, 256, 0, stream>>>(att_out, stat_m, stat_i);

  // 5) P @ V  -> res_bf [B*T, E]   (512 blocks; 8 i-tiles share V via LDS)
  k_pv<<<BQ * HH * 16, 256, 0, stream>>>(att_out, Vb, stat_m, stat_i, res_bf);

  // 6) proj: mha = res @ proj_w^T   (M=4096,N=1024,K=1024)
  k_gemm<<<64 * 32, 256, 0, stream>>>(res_bf, projT, nullptr, mha, nullptr,
                                      BQ * TT, EE, EE, 0);
  // 7) x1 = LN(x + mha)
  k_add_ln<<<BQ * TT, 256, 0, stream>>>(x, mha, ln1_w, ln1_b, x1f, x1bf);

  // 8) h1 = gelu(x1 @ ff_w1^T + b1)   (M=4096,N=4096,K=1024)
  k_gemm<<<256 * 32, 256, 0, stream>>>(x1bf, ff1T, ff_b1, nullptr, h1bf,
                                       BQ * TT, FFN, EE, 1);
  // 9) ff = h1 @ ff_w2^T + b2        (M=4096,N=1024,K=4096)
  k_gemm<<<64 * 32, 256, 0, stream>>>(h1bf, ff2T, ff_b2, fff, nullptr,
                                      BQ * TT, EE, FFN, 0);
  // 10) x2 = LN(x1 + ff) -> d_out
  k_add_ln<<<BQ * TT, 256, 0, stream>>>(x1f, fff, ln2_w, ln2_b, x2_out, nullptr);
}
